// BERTse_76227079569428
// MI455X (gfx1250) — compile-verified, matrix operand order checked by reading, not run on hardware
//
#include <hip/hip_runtime.h>
#include <hip/hip_bf16.h>
#include <math.h>
#include <stdint.h>

// ---------------------------------------------------------------------------
// CDNA5 (gfx1250) wave32 WMMA types
// ---------------------------------------------------------------------------
typedef __attribute__((ext_vector_type(16))) __bf16 v16bf;
typedef __attribute__((ext_vector_type(8)))  __bf16 bf16x8;
typedef __attribute__((ext_vector_type(8)))  float  v8f;

union Frag {
    v16bf  v;
    bf16x8 h[2];
    __bf16 e[16];
};

__device__ __forceinline__ v8f vzero8() {
    v8f z = {0.f, 0.f, 0.f, 0.f, 0.f, 0.f, 0.f, 0.f};
    return z;
}
__device__ __forceinline__ float gelu_f(float x) {
    return 0.5f * x * (1.f + erff(x * 0.70710678118654752f));
}

// Async global->LDS copy of 16 bytes (CDNA5, ASYNCcnt-tracked).
__device__ __forceinline__ void async_b128(unsigned lds_off, const void* gaddr) {
    asm volatile("global_load_async_to_lds_b128 %0, %1, off"
                 :: "v"(lds_off), "v"(gaddr) : "memory");
}
__device__ __forceinline__ void wait_async0() {
    asm volatile("s_wait_asynccnt 0x0" ::: "memory");
}

// ---------------------------------------------------------------------------
// One-time weight convert+transpose: W f32 [K][N] -> Wt bf16 [Npad][K]
// (rows n >= N are zero-filled so the GEMM hot loop needs no N guards)
// ---------------------------------------------------------------------------
__global__ __launch_bounds__(256) void transpose_w_kernel(
    const float* __restrict__ W, __bf16* __restrict__ Wt, int K, int N)
{
    __shared__ __bf16 tile[64][65];
    const int kt = blockIdx.y * 64;
    const int nt = blockIdx.x * 64;
    const int tid = threadIdx.x;
    {
        const int r = tid >> 2;              // k-local 0..63
        const int cg = (tid & 3) * 16;       // n-local base
#pragma unroll
        for (int i = 0; i < 16; i++) {
            const int n = nt + cg + i;
            const float v = (n < N) ? W[(size_t)(kt + r) * N + n] : 0.f;
            tile[cg + i][r] = (__bf16)v;
        }
    }
    __syncthreads();
    {
        const int nl = tid >> 2;             // n-local 0..63
        const int kg = (tid & 3) * 16;       // k-local base
        __bf16* dst = Wt + (size_t)(nt + nl) * K + kt + kg;
#pragma unroll
        for (int i = 0; i < 16; i++) dst[i] = tile[nl][kg + i];
    }
}

// f32 -> bf16 flat convert
__global__ __launch_bounds__(256) void cvt_kernel(
    const float* __restrict__ in, __bf16* __restrict__ out, int n)
{
    const int i = blockIdx.x * 256 + threadIdx.x;
    if (i < n) out[i] = (__bf16)in[i];
}

// ---------------------------------------------------------------------------
// bf16 WMMA GEMM:  C[M,N] = act(A[M,K] @ Wt[N,K]^T + bias) (+resid)
// A bf16 [M][K] row-major, Wt bf16 [Npad][K] (pre-transposed, zero padded).
// Double-buffered LDS tiles filled with async global->LDS copies.
// Block tile 128x128, 8 waves (2x4), wave tile 64x32, K chunk 32.
// act: 0=none 1=gelu 2=sigmoid.  Cf (f32) and/or Cb (bf16) outputs.
// ---------------------------------------------------------------------------
__global__ __launch_bounds__(256) void gemm_bf16_wmma(
    const __bf16* __restrict__ A, const __bf16* __restrict__ Wt,
    const float* __restrict__ bias, const float* __restrict__ resid,
    float* __restrict__ Cf, __bf16* __restrict__ Cb,
    int M, int N, int K, int act)
{
    __shared__ __bf16 As[2][128][40];   // [buf][m][k], 80B row stride
    __shared__ __bf16 Bs[2][128][40];   // [buf][n][k]

    const int tid  = threadIdx.x;
    const int lane = tid & 31;
    const int wave = tid >> 5;
    const int wm   = wave & 1;
    const int wn   = wave >> 1;
    const int g    = lane >> 4;
    const int lr   = lane & 15;
    const int bM   = blockIdx.y * 128;
    const int bN   = blockIdx.x * 128;

    // staging addresses for this thread (row t/2, 32B half-row t&1)
    const int srow = tid >> 1;
    const int shalf = tid & 1;
    const __bf16* aSrc = A  + (size_t)(bM + srow) * K + shalf * 16;
    const __bf16* bSrc = Wt + (size_t)(bN + srow) * K + shalf * 16;
    const unsigned aDst[2] = {
        (unsigned)(uintptr_t)&As[0][srow][shalf * 16],
        (unsigned)(uintptr_t)&As[1][srow][shalf * 16] };
    const unsigned bDst[2] = {
        (unsigned)(uintptr_t)&Bs[0][srow][shalf * 16],
        (unsigned)(uintptr_t)&Bs[1][srow][shalf * 16] };

    v8f acc[4][2];
#pragma unroll
    for (int mi = 0; mi < 4; mi++)
#pragma unroll
        for (int nj = 0; nj < 2; nj++) acc[mi][nj] = vzero8();

    // prologue: stage K-chunk 0 into buffer 0
    async_b128(aDst[0],      aSrc);
    async_b128(aDst[0] + 16, aSrc + 8);
    async_b128(bDst[0],      bSrc);
    async_b128(bDst[0] + 16, bSrc + 8);
    wait_async0();
    __syncthreads();

    for (int kb = 0; kb < K; kb += 32) {
        const int  cur  = (kb >> 5) & 1;
        const bool more = (kb + 32) < K;
        if (more) {  // async-stage next chunk while we compute this one
            const __bf16* an = aSrc + kb + 32;
            const __bf16* bn = bSrc + kb + 32;
            async_b128(aDst[cur ^ 1],      an);
            async_b128(aDst[cur ^ 1] + 16, an + 8);
            async_b128(bDst[cur ^ 1],      bn);
            async_b128(bDst[cur ^ 1] + 16, bn + 8);
        }

        Frag fa[4], fb[2];
#pragma unroll
        for (int mi = 0; mi < 4; mi++) {
            const __bf16* p = &As[cur][wm * 64 + mi * 16 + lr][8 * g];
            fa[mi].h[0] = *(const bf16x8*)p;
            fa[mi].h[1] = *(const bf16x8*)(p + 16);
        }
#pragma unroll
        for (int nj = 0; nj < 2; nj++) {
            const __bf16* p = &Bs[cur][wn * 32 + nj * 16 + lr][8 * g];
            fb[nj].h[0] = *(const bf16x8*)p;
            fb[nj].h[1] = *(const bf16x8*)(p + 16);
        }
#pragma unroll
        for (int mi = 0; mi < 4; mi++)
#pragma unroll
            for (int nj = 0; nj < 2; nj++)
                acc[mi][nj] = __builtin_amdgcn_wmma_f32_16x16x32_bf16(
                    false, fa[mi].v, false, fb[nj].v, (short)0, acc[mi][nj], false, false);

        if (more) wait_async0();
        __syncthreads();
    }

    // epilogue
#pragma unroll
    for (int mi = 0; mi < 4; mi++) {
#pragma unroll
        for (int nj = 0; nj < 2; nj++) {
            const int col = bN + wn * 32 + nj * 16 + lr;
            if (col < N) {
#pragma unroll
                for (int r = 0; r < 8; r++) {
                    const int row = bM + wm * 64 + mi * 16 + r + 8 * g;
                    float v = acc[mi][nj][r] + bias[col];
                    if (act == 1)      v = gelu_f(v);
                    else if (act == 2) v = 1.f / (1.f + __expf(-v));
                    if (resid) v += resid[(size_t)row * N + col];
                    if (Cf) Cf[(size_t)row * N + col] = v;
                    if (Cb) Cb[(size_t)row * N + col] = (__bf16)v;
                }
            }
        }
    }
}

// ---------------------------------------------------------------------------
// Flash attention (bf16 q/k/v in, bf16 ctx out), one block = (b,h,128 q rows).
// K/V key-blocks async-staged to LDS, shared by all 8 waves.
// ---------------------------------------------------------------------------
__global__ __launch_bounds__(256) void attn_kernel(
    const __bf16* __restrict__ q, const __bf16* __restrict__ k,
    const __bf16* __restrict__ v, const float* __restrict__ concept,
    const int* __restrict__ mask, __bf16* __restrict__ ctx, int T, int H)
{
    const int Dm = 1024;
    const int nq = T / 128;
    const int qt = blockIdx.x % nq;
    const int bh = blockIdx.x / nq;
    const int h  = bh % H;
    const int b  = bh / H;
    const int qb0 = qt * 128;

    __shared__ __bf16 Qs[128][72];    // q tile
    __shared__ __bf16 Ks[32][72];     // key block [key][d]
    __shared__ __bf16 Vr[32][72];     // value block [key][d]
    __shared__ __bf16 Pw[8][16][40];  // per-wave P bounce

    const int tid  = threadIdx.x;
    const int lane = tid & 31;
    const int wave = tid >> 5;
    const int g    = lane >> 4;
    const int lr   = lane & 15;

    const __bf16* qbase = q + (size_t)b * T * Dm + h * 64;
    const __bf16* kbase = k + (size_t)b * T * Dm + h * 64;
    const __bf16* vbase = v + (size_t)b * T * Dm + h * 64;

    // async-stage Q tile: 128 rows x 128B
    {
        const int row = tid >> 1, half = tid & 1;
        const __bf16* src = qbase + (size_t)(qb0 + row) * Dm + half * 32;
        const unsigned dst = (unsigned)(uintptr_t)&Qs[row][half * 32];
        async_b128(dst,      src);
        async_b128(dst + 16, src + 8);
        async_b128(dst + 32, src + 16);
        async_b128(dst + 48, src + 24);
    }
    wait_async0();
    __syncthreads();

    const int qrow_base = qb0 + wave * 16;
    v8f o[4];
#pragma unroll
    for (int nj = 0; nj < 4; nj++) o[nj] = vzero8();
    float mst[8], lst[8];
#pragma unroll
    for (int r = 0; r < 8; r++) { mst[r] = -1e30f; lst[r] = 0.f; }

    // per-thread K/V staging slot: threads 0..127 -> Ks, 128..255 -> Vr
    const int su   = tid & 127;
    const int srow = su >> 2;            // key-local 0..31
    const int schk = (su & 3) * 16;      // 32B chunk (16 bf16)
    const __bf16* sbase = (tid < 128 ? kbase : vbase);
    const unsigned sdst  = (tid < 128)
        ? (unsigned)(uintptr_t)&Ks[srow][schk]
        : (unsigned)(uintptr_t)&Vr[srow][schk];

    for (int kbk = 0; kbk < T; kbk += 32) {
        // stage K/V key block
        {
            const __bf16* src = sbase + (size_t)(kbk + srow) * Dm + schk;
            async_b128(sdst,      src);
            async_b128(sdst + 16, src + 8);
        }
        wait_async0();
        __syncthreads();

        // ---- S = Q @ K^T over dk=64 (two K=32 chunks) ----
        v8f s0 = vzero8(), s1 = vzero8();
#pragma unroll
        for (int kc = 0; kc < 2; kc++) {
            Frag fqa, fk0, fk1;
            {
                const __bf16* p = &Qs[wave * 16 + lr][kc * 32 + 8 * g];
                fqa.h[0] = *(const bf16x8*)p;
                fqa.h[1] = *(const bf16x8*)(p + 16);
            }
            {
                const __bf16* p = &Ks[lr][kc * 32 + 8 * g];
                fk0.h[0] = *(const bf16x8*)p;
                fk0.h[1] = *(const bf16x8*)(p + 16);
            }
            {
                const __bf16* p = &Ks[16 + lr][kc * 32 + 8 * g];
                fk1.h[0] = *(const bf16x8*)p;
                fk1.h[1] = *(const bf16x8*)(p + 16);
            }
            s0 = __builtin_amdgcn_wmma_f32_16x16x32_bf16(false, fqa.v, false, fk0.v, (short)0, s0, false, false);
            s1 = __builtin_amdgcn_wmma_f32_16x16x32_bf16(false, fqa.v, false, fk1.v, (short)0, s1, false, false);
        }

        // ---- scale + concept bias + mask ----
        const int colk0 = kbk + lr;
        const int colk1 = kbk + 16 + lr;
        const float mk0 = (mask[b * T + colk0] == 0) ? -1e9f : 0.f;
        const float mk1 = (mask[b * T + colk1] == 0) ? -1e9f : 0.f;
#pragma unroll
        for (int r = 0; r < 8; r++) {
            const int qg = qrow_base + r + 8 * g;
            float b0 = mk0, b1 = mk1;
            if (h < 8) {
                const size_t cb = (((size_t)b * T + qg) * T) * 8 + h;
                b0 += 5.0f * concept[cb + (size_t)colk0 * 8];
                b1 += 5.0f * concept[cb + (size_t)colk1 * 8];
            }
            s0[r] = s0[r] * 0.125f + b0;   // 1/sqrt(dk)
            s1[r] = s1[r] * 0.125f + b1;
        }

        // ---- online softmax ----
#pragma unroll
        for (int r = 0; r < 8; r++) {
            float bm = fmaxf(s0[r], s1[r]);
            bm = fmaxf(bm, __shfl_xor(bm, 1));
            bm = fmaxf(bm, __shfl_xor(bm, 2));
            bm = fmaxf(bm, __shfl_xor(bm, 4));
            bm = fmaxf(bm, __shfl_xor(bm, 8));
            const float nm = fmaxf(mst[r], bm);
            const float sc = __expf(mst[r] - nm);
            const float p0 = __expf(s0[r] - nm);
            const float p1 = __expf(s1[r] - nm);
            float ps = p0 + p1;
            ps += __shfl_xor(ps, 1);
            ps += __shfl_xor(ps, 2);
            ps += __shfl_xor(ps, 4);
            ps += __shfl_xor(ps, 8);
            lst[r] = lst[r] * sc + ps;
            mst[r] = nm;
            o[0][r] *= sc; o[1][r] *= sc; o[2][r] *= sc; o[3][r] *= sc;
            Pw[wave][r + 8 * g][lr]      = (__bf16)p0;
            Pw[wave][r + 8 * g][16 + lr] = (__bf16)p1;
        }

        // ---- O += P @ V  (K=32 keys, V fragments from LDS) ----
        Frag fpa;
        {
            const __bf16* p = &Pw[wave][lr][8 * g];
            fpa.h[0] = *(const bf16x8*)p;
            fpa.h[1] = *(const bf16x8*)(p + 16);
        }
#pragma unroll
        for (int nj = 0; nj < 4; nj++) {
            Frag fv;
            const int dcol = nj * 16 + lr;
#pragma unroll
            for (int e = 0; e < 8; e++) {
                fv.e[e]     = Vr[8 * g + e][dcol];
                fv.e[e + 8] = Vr[16 + 8 * g + e][dcol];
            }
            o[nj] = __builtin_amdgcn_wmma_f32_16x16x32_bf16(false, fpa.v, false, fv.v, (short)0, o[nj], false, false);
        }
        __syncthreads();   // all waves done with Ks/Vr before next stage
    }

    // ---- normalize + write ctx (bf16) ----
#pragma unroll
    for (int r = 0; r < 8; r++) {
        const float inv = 1.0f / fmaxf(lst[r], 1e-20f);
        const int qg = qrow_base + r + 8 * g;
        __bf16* cp = ctx + (size_t)(b * T + qg) * Dm + h * 64;
#pragma unroll
        for (int nj = 0; nj < 4; nj++) cp[nj * 16 + lr] = (__bf16)(o[nj][r] * inv);
    }
}

// ---------------------------------------------------------------------------
// Row LayerNorm: f32 in (+ optional pos/seg add), f32 and/or bf16 out.
// ---------------------------------------------------------------------------
__global__ __launch_bounds__(256) void ln_kernel(
    const float* __restrict__ x, float* __restrict__ outf, __bf16* __restrict__ outb,
    const float* __restrict__ gamma, const float* __restrict__ beta,
    const float* __restrict__ pos_emb, const float* __restrict__ seg_emb,
    const int* __restrict__ segids, int addmode, int T)
{
    const int Dm = 1024;
    const int row = blockIdx.x;
    const float* xr = x + (size_t)row * Dm;
    const float* av = nullptr;
    if (addmode == 1)      av = pos_emb + (size_t)(row % T) * Dm;
    else if (addmode == 2) av = seg_emb + (size_t)segids[row] * Dm;

    const int tid = threadIdx.x, lane = tid & 31, wave = tid >> 5;
    float vals[4];
    float s = 0.f;
#pragma unroll
    for (int i = 0; i < 4; i++) {
        const int c = tid + 256 * i;
        float vv = xr[c];
        if (av) vv += av[c];
        vals[i] = vv;
        s += vv;
    }
    __shared__ float red[8];
#pragma unroll
    for (int m = 16; m >= 1; m >>= 1) s += __shfl_xor(s, m);
    if (lane == 0) red[wave] = s;
    __syncthreads();
    float tot = 0.f;
#pragma unroll
    for (int i = 0; i < 8; i++) tot += red[i];
    const float mean = tot * (1.f / 1024.f);
    __syncthreads();

    float vs = 0.f;
#pragma unroll
    for (int i = 0; i < 4; i++) { const float d = vals[i] - mean; vs += d * d; }
#pragma unroll
    for (int m = 16; m >= 1; m >>= 1) vs += __shfl_xor(vs, m);
    if (lane == 0) red[wave] = vs;
    __syncthreads();
    float vtot = 0.f;
#pragma unroll
    for (int i = 0; i < 8; i++) vtot += red[i];
    const float rstd = rsqrtf(vtot * (1.f / 1024.f) + 1e-12f);

#pragma unroll
    for (int i = 0; i < 4; i++) {
        const int c = tid + 256 * i;
        const float y = gamma[c] * (vals[i] - mean) * rstd + beta[c];
        if (outf) outf[(size_t)row * Dm + c] = y;
        if (outb) outb[(size_t)row * Dm + c] = (__bf16)y;
    }
}

// ---------------------------------------------------------------------------
// Highway combine: x = T*H + (1-T)*x  -> f32 X and bf16 copy
// ---------------------------------------------------------------------------
__global__ __launch_bounds__(256) void highway_kernel(
    const float* __restrict__ Tx, const float* __restrict__ Hx,
    const float* __restrict__ xin, float* __restrict__ xout,
    __bf16* __restrict__ xbf, int n)
{
    const int i = blockIdx.x * 256 + threadIdx.x;
    if (i < n) {
        const float t = Tx[i];
        const float v = t * Hx[i] + (1.f - t) * xin[i];
        xout[i] = v;
        xbf[i]  = (__bf16)v;
    }
}

// ---------------------------------------------------------------------------
// Orchestration
// ---------------------------------------------------------------------------
extern "C" void kernel_launch(void* const* d_in, const int* in_sizes, int n_in,
                              void* d_out, int out_size, void* d_ws, size_t ws_size,
                              hipStream_t stream)
{
    (void)in_sizes; (void)n_in; (void)out_size; (void)ws_size;
    constexpr int Bb = 4, Tt = 1024, Dd = 1024, DFFd = 4096, Hh = 16;
    const int M = Bb * Tt;

    const float* qa      = (const float*)d_in[0];
    const int*   segids  = (const int*)d_in[1];
    const int*   maskp   = (const int*)d_in[2];
    const float* concept = (const float*)d_in[3];
    auto F = [&](int i) { return (const float*)d_in[i]; };

    // params flattened in sorted-key pytree order
    const float *cls_b = F(4), *cls_w = F(5);
    const float *fnb = F(6), *fnw = F(7);
    const float *fcb = F(8), *fcw = F(9);
    const float *g0b = F(10), *g0w = F(11), *g1b = F(12), *g1w = F(13);
    const float *l0b = F(14), *l0w = F(15), *l1b = F(16), *l1w = F(17);
    const float *pos_emb = F(114), *pnb = F(115), *pnw = F(116);
    const float *seg_emb = F(117), *snb = F(118), *snw = F(119);

    // ---- workspace layout ----
    char* ws = (char*)d_ws;
    size_t off = 0;
    auto allocF = [&](size_t elems) { float* p = (float*)(ws + off); off += ((elems * 4 + 255) & ~(size_t)255); return p; };
    auto allocB = [&](size_t elems) { __bf16* p = (__bf16*)(ws + off); off += ((elems * 2 + 255) & ~(size_t)255); return p; };

    float*  X    = allocF((size_t)M * Dd);       // residual stream (f32)
    float*  TAf  = allocF((size_t)M * Dd);       // f32 temp (fc out / highway H)
    float*  TCf  = allocF((size_t)M * Dd);       // f32 temp (highway T)
    __bf16* Xbf  = allocB((size_t)M * Dd);       // bf16 copy of X
    __bf16* TAb  = allocB((size_t)M * Dd);       // LN out / ctx (bf16)
    __bf16* TBb  = allocB((size_t)M * DFFd);     // ff1 out (bf16)
    __bf16* Qb   = allocB((size_t)M * Dd);
    __bf16* Kb   = allocB((size_t)M * Dd);
    __bf16* Vb   = allocB((size_t)M * Dd);
    __bf16* QAbf = allocB((size_t)M * Dd);       // bf16 copy of qa

    // transposed bf16 weights, [Npad][K]
    __bf16 *Wl0 = allocB((size_t)Dd * Dd), *Wl1 = allocB((size_t)Dd * Dd);
    __bf16 *Wg0 = allocB((size_t)Dd * Dd), *Wg1 = allocB((size_t)Dd * Dd);
    __bf16 *Wfc = allocB((size_t)Dd * Dd);
    __bf16 *Wcls = allocB((size_t)128 * Dd);
    struct LW { __bf16 *q, *k, *v, *o, *f1, *f2; } Lw[6];
    for (int i = 0; i < 6; i++) {
        Lw[i].q  = allocB((size_t)Dd * Dd);
        Lw[i].k  = allocB((size_t)Dd * Dd);
        Lw[i].v  = allocB((size_t)Dd * Dd);
        Lw[i].o  = allocB((size_t)Dd * Dd);
        Lw[i].f1 = allocB((size_t)DFFd * Dd);    // [N=4096][K=1024]
        Lw[i].f2 = allocB((size_t)Dd * DFFd);    // [N=1024][K=4096]
    }

    dim3 blk(256);
    auto trans = [&](const float* W, __bf16* Wt, int K, int N, int Npad) {
        transpose_w_kernel<<<dim3(Npad / 64, K / 64), blk, 0, stream>>>(W, Wt, K, N);
    };
    auto gemm = [&](const __bf16* A, const __bf16* Wt, const float* bias,
                    const float* resid, float* Cf, __bf16* Cb, int N, int K, int act) {
        const int Npad = (N + 127) & ~127;
        gemm_bf16_wmma<<<dim3(Npad / 128, M / 128), blk, 0, stream>>>(
            A, Wt, bias, resid, Cf, Cb, M, N, K, act);
    };
    auto ln = [&](const float* x, float* of, __bf16* ob,
                  const float* gw, const float* gb, int mode) {
        ln_kernel<<<dim3(M), blk, 0, stream>>>(x, of, ob, gw, gb, pos_emb, seg_emb, segids, mode, Tt);
    };

    // ---- one-time weight convert + transpose, input convert ----
    trans(l0w, Wl0, Dd, Dd, Dd);  trans(l1w, Wl1, Dd, Dd, Dd);
    trans(g0w, Wg0, Dd, Dd, Dd);  trans(g1w, Wg1, Dd, Dd, Dd);
    trans(fcw, Wfc, Dd, Dd, Dd);  trans(cls_w, Wcls, Dd, 3, 128);
    for (int i = 0; i < 6; i++) {
        const int p = 18 + i * 16;
        trans(F(p + 1),  Lw[i].f1, Dd, DFFd, DFFd);   // ff1.w
        trans(F(p + 3),  Lw[i].f2, DFFd, Dd, Dd);     // ff2.w
        trans(F(p + 5),  Lw[i].k,  Dd, Dd, Dd);       // k.w
        trans(F(p + 11), Lw[i].o,  Dd, Dd, Dd);       // o.w
        trans(F(p + 13), Lw[i].q,  Dd, Dd, Dd);       // q.w
        trans(F(p + 15), Lw[i].v,  Dd, Dd, Dd);       // v.w
    }
    cvt_kernel<<<dim3((M * Dd) / 256), blk, 0, stream>>>(qa, QAbf, M * Dd);

    // ---- highway layers ----
    gemm(QAbf, Wl0, l0b, nullptr, TAf, nullptr, Dd, Dd, 1);   // Hx = gelu
    gemm(QAbf, Wg0, g0b, nullptr, TCf, nullptr, Dd, Dd, 2);   // Tx = sigmoid
    highway_kernel<<<dim3((M * Dd) / 256), blk, 0, stream>>>(TCf, TAf, qa, X, Xbf, M * Dd);
    gemm(Xbf, Wl1, l1b, nullptr, TAf, nullptr, Dd, Dd, 1);
    gemm(Xbf, Wg1, g1b, nullptr, TCf, nullptr, Dd, Dd, 2);
    highway_kernel<<<dim3((M * Dd) / 256), blk, 0, stream>>>(TCf, TAf, X, X, Xbf, M * Dd);

    // ---- fc + pos/seg embeddings ----
    gemm(Xbf, Wfc, fcb, nullptr, TAf, nullptr, Dd, Dd, 0);
    ln(TAf, X, nullptr, pnw, pnb, 1);
    ln(X,   X, nullptr, snw, snb, 2);

    // ---- transformer layers ----
    for (int li = 0; li < 6; li++) {
        const int p = 18 + li * 16;
        const float *ff1b = F(p + 0), *ff2b = F(p + 2);
        const float *kb_  = F(p + 4), *nib = F(p + 6), *niw = F(p + 7);
        const float *nob  = F(p + 8), *now_ = F(p + 9);
        const float *ob   = F(p + 10), *qb_ = F(p + 12), *vb_ = F(p + 14);

        ln(X, nullptr, TAb, niw, nib, 0);                       // res = LN(x) (bf16)
        gemm(TAb, Lw[li].q, qb_, nullptr, nullptr, Qb, Dd, Dd, 0);
        gemm(TAb, Lw[li].k, kb_, nullptr, nullptr, Kb, Dd, Dd, 0);
        gemm(TAb, Lw[li].v, vb_, nullptr, nullptr, Vb, Dd, Dd, 0);
        attn_kernel<<<dim3(Bb * Hh * (Tt / 128)), blk, 0, stream>>>(
            Qb, Kb, Vb, concept, maskp, TAb, Tt, Hh);           // ctx -> TAb (bf16)
        gemm(TAb, Lw[li].o, ob, X, X, nullptr, Dd, Dd, 0);      // x += ctx@Wo+bo
        ln(X, nullptr, TAb, now_, nob, 0);
        gemm(TAb, Lw[li].f1, ff1b, nullptr, nullptr, TBb, DFFd, Dd, 1);
        gemm(TBb, Lw[li].f2, ff2b, X, X, nullptr, Dd, DFFd, 0); // x += ff
    }

    // ---- final norm + classifier ----
    ln(X, nullptr, TAb, fnw, fnb, 0);
    gemm(TAb, Wcls, cls_b, nullptr, (float*)d_out, nullptr, 3, Dd, 0);
}
